// ScaleVQ_63866163692080
// MI455X (gfx1250) — matrix operand order repaired and checked by reading, not compiled, not run on hardware
//
#include <hip/hip_runtime.h>
#include <hip/hip_bf16.h>

typedef __attribute__((ext_vector_type(16))) _Float16 v16h;
typedef __attribute__((ext_vector_type(8)))  _Float16 v8h;
typedef __attribute__((ext_vector_type(8)))  float    v8f;

#define BQ 256
#define CQ 512
#define TQ 64
#define TP 66          // T + 2 halo rows (t=-1, t=64) for branch-free conv loads
#define VQ 1024
#define QQ 7
static constexpr size_t NZ = (size_t)BQ * CQ * TQ;   // 8,388,608

// ---------------------------------------------------------------------------
// WMMA fragment helpers (wave32, v_wmma_f32_16x16x32_f16 operands in this file's contiguous-K order, A and B alike)
// A (16x32 f16): lane L holds row M=L&15; K-base = (L>=16)?16:0; halves 0..15 at
// K=kb..kb+15 (contiguous, matches B). Works for global or LDS ptr.
// ---------------------------------------------------------------------------
__device__ inline v16h load_afrag(const _Float16* base, int ld, int k0) {
  const int lane = threadIdx.x & 31;
  const int mrow = lane & 15;
  const int kb   = (lane >> 4) << 4;              // 0 or 16 halves
  const _Float16* p = base + (size_t)mrow * ld + k0;
  union { v16h v; v8h h[2]; } u;
  u.h[0] = *(const v8h*)(p + kb);
  u.h[1] = *(const v8h*)(p + kb + 8);
  return u.v;
}

__device__ inline v16h load_bfrag(const _Float16* __restrict__ base, int ld, int k0) {
  const int lane = threadIdx.x & 31;
  const int ncol = lane & 15;
  const int kb   = (lane >> 4) << 4;              // 0 or 16 halves
  return *(const v16h*)(base + (size_t)ncol * ld + k0 + kb);
}

__device__ inline v8f wmma_f16(v16h a, v16h b, v8f c) {
  return __builtin_amdgcn_wmma_f32_16x16x32_f16(false, a, false, b, (short)0, c,
                                                false, false);
}

// ---------------------------------------------------------------------------
__global__ void init_kernel(const float* __restrict__ z, float* __restrict__ zq,
                            float* __restrict__ zhat) {
  size_t g = (size_t)blockIdx.x * blockDim.x + threadIdx.x;
  if (g < NZ) { zq[g] = z[g]; zhat[g] = 0.0f; }
}

__global__ void zero_scalars_kernel(float* __restrict__ accs) {
  if (blockIdx.x == 0 && threadIdx.x < 32) accs[threadIdx.x] = 0.0f;
}

// zero the halo rows (t=-1 and t=64) of both padded activation buffers; the
// interior is rewritten every scale, the halo stays zero for the whole call.
__global__ void zero_pads_kernel(_Float16* __restrict__ ut,
                                 _Float16* __restrict__ h1) {
  int g = blockIdx.x * blockDim.x + threadIdx.x;  // b*C + c
  int c = g & (CQ - 1);
  int b = g >> 9;
  size_t lo = (size_t)b * TP * CQ + c;
  size_t hi = lo + (size_t)(TP - 1) * CQ;
  ut[lo] = (_Float16)0.f; ut[hi] = (_Float16)0.f;
  h1[lo] = (_Float16)0.f; h1[hi] = (_Float16)0.f;
}

// Codebooks -> f16 copy + per-code squared norms. One block per (q,v) row.
__global__ void cb_prep_kernel(const float* __restrict__ cb,
                               _Float16* __restrict__ cbh,
                               float* __restrict__ cbnorm) {
  const int rowid = blockIdx.x;                   // q*V + v
  const float* src = cb + (size_t)rowid * CQ;
  _Float16* dst = cbh + (size_t)rowid * CQ;
  int t = threadIdx.x;
  float a0 = src[t], a1 = src[t + 256];
  dst[t] = (_Float16)a0;
  dst[t + 256] = (_Float16)a1;
  float s = a0 * a0 + a1 * a1;
  for (int o = 16; o > 0; o >>= 1) s += __shfl_down(s, o, 32);
  __shared__ float red[8];
  if ((threadIdx.x & 31) == 0) red[threadIdx.x >> 5] = s;
  __syncthreads();
  if (threadIdx.x == 0) {
    float tot = 0.f;
    for (int i = 0; i < 8; ++i) tot += red[i];
    cbnorm[rowid] = tot;
  }
}

// phi_w (Q,2,Co,Ci,3) f32 -> wre [(q,d,dk)][Co][Ci] f16 (K=Ci contiguous)
__global__ void convert_w_kernel(const float* __restrict__ w,
                                 _Float16* __restrict__ wre) {
  size_t g = (size_t)blockIdx.x * blockDim.x + threadIdx.x;
  const size_t TOT = (size_t)QQ * 2 * 3 * CQ * CQ;
  if (g >= TOT) return;
  int ci = (int)(g & (CQ - 1));
  size_t t1 = g >> 9;
  int co = (int)(t1 & (CQ - 1));
  size_t t2 = t1 >> 9;
  int dk = (int)(t2 % 3);
  int qd = (int)(t2 / 3);
  size_t src = (((size_t)qd * CQ + co) * CQ + ci) * 3 + dk;
  wre[g] = (_Float16)w[src];
}

// area_down + transpose + f32->f16: tok[(b*s+j)][c] = mean_r z_q[b,c,j*r..]
__global__ void pool_tok_kernel(const float* __restrict__ zq,
                                _Float16* __restrict__ tokh, int s, int r) {
  size_t g = (size_t)blockIdx.x * blockDim.x + threadIdx.x;   // m*C + c
  int c = (int)(g & (CQ - 1));
  int m = (int)(g >> 9);
  int b = m / s, j = m - b * s;
  const float* p = zq + ((size_t)b * CQ + c) * TQ + (size_t)j * r;
  float sum = 0.f;
  for (int t = 0; t < r; ++t) sum += p[t];
  tokh[g] = (_Float16)(sum / (float)r);
}

__global__ void zero_counts_kernel(int* __restrict__ counts) {
  int g = blockIdx.x * blockDim.x + threadIdx.x;
  if (g < VQ) counts[g] = 0;
}

// ---------------------------------------------------------------------------
// Fused distance-GEMM + argmax. Block = 16-row M tile; 8 waves x 8 N-tiles
// cover all V=1024 codes. Token A-tile staged once in LDS (16 KB) and read
// via ds_read, so VMEM sched groups contain exactly the codebook B loads.
// Two K-sweeps of 4 N-tiles keep accumulators at 32 VGPRs (no spills).
// Cadence per sweep: V8,D2 | {V8, D2, M4} x15 | M4 (skew 1 chunk).
// ---------------------------------------------------------------------------
__global__ void score_argmax_kernel(const _Float16* __restrict__ tokh,
                                    const _Float16* __restrict__ cbh,
                                    const float* __restrict__ cbnorm,
                                    int* __restrict__ idx_int,
                                    float* __restrict__ idx_out,
                                    int* __restrict__ counts) {
  const int wave = threadIdx.x >> 5;
  const int lane = threadIdx.x & 31;
  const int mtile = blockIdx.x;
  const int ncol = lane & 15;
  const _Float16* Bbase = cbh + (size_t)wave * 8 * 16 * CQ;

  __shared__ _Float16 Alds[16 * CQ];              // 16 KB token tile
  {
    const int tid = threadIdx.x;
    const v8h* src = (const v8h*)(tokh + (size_t)mtile * 16 * CQ);
    v8h* dst = (v8h*)Alds;
#pragma unroll
    for (int i = 0; i < 4; ++i) dst[tid + 256 * i] = src[tid + 256 * i];
  }
  __syncthreads();
  __builtin_amdgcn_sched_barrier(0);

  float bv[8];
  int bi[8];
#pragma unroll
  for (int i = 0; i < 8; ++i) { bv[i] = -3.4e38f; bi[i] = 0x7fffffff; }

#pragma unroll
  for (int half = 0; half < 2; ++half) {
    const _Float16* Bb = Bbase + (size_t)half * 4 * 16 * CQ;
    v8f acc[4] = {};
#pragma unroll
    for (int kc = 0; kc < 16; ++kc) {
      const int k0 = kc * 32;
      v16h a = load_afrag(Alds, CQ, k0);          // ds_read x2
      v16h bfr[4];
#pragma unroll
      for (int nt = 0; nt < 4; ++nt)
        bfr[nt] = load_bfrag(Bb + (size_t)nt * 16 * CQ, CQ, k0);
#pragma unroll
      for (int nt = 0; nt < 4; ++nt)
        acc[nt] = wmma_f16(a, bfr[nt], acc[nt]);
    }
    // 0x20=VMEM-read (B only), 0x100=DS-read (A), 0x8=MFMA/WMMA
    __builtin_amdgcn_sched_group_barrier(0x020, 8, 0);
    __builtin_amdgcn_sched_group_barrier(0x100, 2, 0);
#pragma unroll
    for (int i = 0; i < 15; ++i) {
      __builtin_amdgcn_sched_group_barrier(0x020, 8, 0);
      __builtin_amdgcn_sched_group_barrier(0x100, 2, 0);
      __builtin_amdgcn_sched_group_barrier(0x008, 4, 0);
    }
    __builtin_amdgcn_sched_group_barrier(0x008, 4, 0);
    __builtin_amdgcn_sched_barrier(0);   // fence argmax fold out of the groups

    // fold this half's 4 tiles into the running per-row argmax
#pragma unroll
    for (int nt = 0; nt < 4; ++nt) {
      int n = (wave * 8 + half * 4 + nt) * 16 + ncol;
      float bias = 0.5f * cbnorm[n];
#pragma unroll
      for (int i = 0; i < 8; ++i) {
        float v = acc[nt][i] - bias;
        if (v > bv[i] || (v == bv[i] && n < bi[i])) { bv[i] = v; bi[i] = n; }
      }
    }
  }

  // reduce over the 16 lanes that share this row-half (columns differ)
  for (int o = 1; o < 16; o <<= 1) {
#pragma unroll
    for (int i = 0; i < 8; ++i) {
      float ov = __shfl_xor(bv[i], o, 32);
      int   oi = __shfl_xor(bi[i], o, 32);
      if (ov > bv[i] || (ov == bv[i] && oi < bi[i])) { bv[i] = ov; bi[i] = oi; }
    }
  }
  __shared__ float sv[8][16];
  __shared__ int   si[8][16];
  if (ncol == 0) {
    int rbase = (lane >> 4) << 3;   // rows 0..7 (lane 0) or 8..15 (lane 16)
#pragma unroll
    for (int i = 0; i < 8; ++i) { sv[wave][rbase + i] = bv[i]; si[wave][rbase + i] = bi[i]; }
  }
  __syncthreads();
  if (threadIdx.x < 16) {
    int row = threadIdx.x;
    float best = sv[0][row];
    int bidx = si[0][row];
    for (int w = 1; w < 8; ++w) {
      float v = sv[w][row]; int ii = si[w][row];
      if (v > best || (v == best && ii < bidx)) { best = v; bidx = ii; }
    }
    int m = mtile * 16 + row;
    idx_int[m] = bidx;
    idx_out[m] = (float)bidx;
    atomicAdd(&counts[bidx], 1);
  }
}

// gather codebook rows (= straight-through forward value) + linear_up,
// written f16 into padded (B, T+2, C) layout (interior rows t+1).
__global__ void gather_up_kernel(const float* __restrict__ cbq,
                                 const int* __restrict__ idx,
                                 _Float16* __restrict__ ut, int s) {
  size_t g = (size_t)blockIdx.x * blockDim.x + threadIdx.x;  // (b*T+t)*C + c
  int c = (int)(g & (CQ - 1));
  int t = (int)((g >> 9) & (TQ - 1));
  int b = (int)(g >> 15);
  float pos = (t + 0.5f) * ((float)s * (1.0f / TQ)) - 0.5f;
  float fi = floorf(pos);
  float frac = pos - fi;
  int i0 = (int)fi, i1 = (int)fi + 1;
  i0 = i0 < 0 ? 0 : (i0 > s - 1 ? s - 1 : i0);
  i1 = i1 < 0 ? 0 : (i1 > s - 1 ? s - 1 : i1);
  int v0 = idx[b * s + i0];
  int v1 = idx[b * s + i1];
  float val = cbq[(size_t)v0 * CQ + c] * (1.0f - frac)
            + cbq[(size_t)v1 * CQ + c] * frac;
  ut[((size_t)b * TP + t + 1) * CQ + c] = (_Float16)val;
}

// ---------------------------------------------------------------------------
// Conv1d(C,C,k=3,pad=1)+bias+ReLU as 3 shifted 512x512x(B*T) WMMA GEMMs,
// flattened to 48 K-chunks. Weight slab (3x16x512 f16 = 48 KB) staged in LDS
// and shared by all 8 waves; activations (per-wave b) loaded from global with
// halo padding (branch-free). Cadence: V8,D2 | {V8, D2, M4} x47 | M4.
// Wave computes a 16(co) x 64(t, one full b) tile. FINAL fuses z_hat/z_q.
// ---------------------------------------------------------------------------
template <bool FINAL>
__global__ void conv_gemm_kernel(const _Float16* __restrict__ W3,   // [3][C][C]
                                 const _Float16* __restrict__ U,    // padded
                                 const float* __restrict__ bias,
                                 _Float16* __restrict__ Yh,         // !FINAL out
                                 float* __restrict__ zhat,          // FINAL
                                 float* __restrict__ zq) {          // FINAL
  const int wave = threadIdx.x >> 5;
  const int lane = threadIdx.x & 31;
  const int mtile = blockIdx.x;                   // 0..31 (co tiles)
  const int b = blockIdx.y * 8 + wave;            // 0..255
  const int ncol = lane & 15;
  const int kbB  = (lane >> 4) << 4;
  const _Float16* Ub = U + ((size_t)b * TP + 1) * CQ;   // row t = 0

  __shared__ _Float16 Wlds[3 * 16 * CQ];          // 48 KB weight slab
  {
    const int tid = threadIdx.x;
#pragma unroll
    for (int dk = 0; dk < 3; ++dk) {
      const v8h* src = (const v8h*)(W3 + (size_t)dk * CQ * CQ + (size_t)mtile * 16 * CQ);
      v8h* dst = (v8h*)(Wlds + (size_t)dk * 16 * CQ);
#pragma unroll
      for (int i = 0; i < 4; ++i) dst[tid + 256 * i] = src[tid + 256 * i];
    }
  }
  __syncthreads();
  __builtin_amdgcn_sched_barrier(0);

  v8f acc[4] = {};
#pragma unroll
  for (int kk = 0; kk < 48; ++kk) {
    const int dk = kk >> 4;
    const int k0 = (kk & 15) << 5;
    v16h a = load_afrag(Wlds + (size_t)dk * 16 * CQ, CQ, k0);  // ds_read x2
    v16h bb[4];
#pragma unroll
    for (int j = 0; j < 4; ++j) {
      const long tp = j * 16 + ncol + dk - 1;     // in [-1, 64]: halo handles it
      bb[j] = *(const v16h*)(Ub + tp * CQ + k0 + kbB);
    }
#pragma unroll
    for (int j = 0; j < 4; ++j)
      acc[j] = wmma_f16(a, bb[j], acc[j]);
  }
  __builtin_amdgcn_sched_group_barrier(0x020, 8, 0);
  __builtin_amdgcn_sched_group_barrier(0x100, 2, 0);
#pragma unroll
  for (int i = 0; i < 47; ++i) {
    __builtin_amdgcn_sched_group_barrier(0x020, 8, 0);
    __builtin_amdgcn_sched_group_barrier(0x100, 2, 0);
    __builtin_amdgcn_sched_group_barrier(0x008, 4, 0);
  }
  __builtin_amdgcn_sched_group_barrier(0x008, 4, 0);
  __builtin_amdgcn_sched_barrier(0);

  const int mb = (lane >> 4) << 3;                // 0 or 8
  const int co0 = mtile * 16 + mb;
#pragma unroll
  for (int j = 0; j < 4; ++j) {
    int t = j * 16 + ncol;
    if constexpr (FINAL) {
#pragma unroll
      for (int i = 0; i < 8; ++i) {
        float v = acc[j][i] + bias[co0 + i];
        v = v > 0.f ? v : 0.f;
        size_t o = ((size_t)b * CQ + (size_t)(co0 + i)) * TQ + t;
        zhat[o] = zhat[o] + v;                    // z_hat += z_hat_q
        zq[o]   = zq[o] - v;                      // z_q   -= z_hat_q
      }
    } else {
      union { v8h v; _Float16 e[8]; } pk;
#pragma unroll
      for (int i = 0; i < 8; ++i) {
        float v = acc[j][i] + bias[co0 + i];
        pk.e[i] = (_Float16)(v > 0.f ? v : 0.f);
      }
      *(v8h*)(Yh + ((size_t)b * TP + t + 1) * CQ + co0) = pk.v;
    }
  }
}

__global__ void loss_reduce_kernel(const float* __restrict__ zhat,
                                   const float* __restrict__ z,
                                   float* __restrict__ dst) {
  float s = 0.f;
  for (size_t g = (size_t)blockIdx.x * blockDim.x + threadIdx.x; g < NZ;
       g += (size_t)gridDim.x * blockDim.x) {
    float d = zhat[g] - z[g];
    s += d * d;
  }
  for (int o = 16; o > 0; o >>= 1) s += __shfl_down(s, o, 32);
  __shared__ float red[8];
  if ((threadIdx.x & 31) == 0) red[threadIdx.x >> 5] = s;
  __syncthreads();
  if (threadIdx.x == 0) {
    float tot = 0.f;
    for (int i = 0; i < 8; ++i) tot += red[i];
    atomicAdd(dst, tot);
  }
}

__global__ void perp_usage_kernel(const int* __restrict__ counts, int M,
                                  float* __restrict__ usage_dst,
                                  float* __restrict__ perp_dst) {
  float ent = 0.f, used = 0.f;
  for (int v = threadIdx.x; v < VQ; v += blockDim.x) {
    float c = (float)counts[v];
    float p = c / (float)M;
    ent += p * logf(p + 1e-10f);
    if (c > 0.f) used += 1.f;
  }
  for (int o = 16; o > 0; o >>= 1) {
    ent += __shfl_down(ent, o, 32);
    used += __shfl_down(used, o, 32);
  }
  __shared__ float re[8], ru[8];
  if ((threadIdx.x & 31) == 0) { re[threadIdx.x >> 5] = ent; ru[threadIdx.x >> 5] = used; }
  __syncthreads();
  if (threadIdx.x == 0) {
    float e = 0.f, u = 0.f;
    for (int i = 0; i < 8; ++i) { e += re[i]; u += ru[i]; }
    *perp_dst = expf(-e);
    *usage_dst = u / (float)VQ * 100.0f;
  }
}

// accs layout: [0..6]=sum((zhat-z)^2) per q, [7..13]=usage_q, [14..20]=perp_q
__global__ void finalize_kernel(const float* __restrict__ accs,
                                float* __restrict__ out_scalars) {
  if (threadIdx.x != 0 || blockIdx.x != 0) return;
  float loss = 0.f, usage = 0.f, perp = 0.f;
  for (int q = 0; q < QQ; ++q) {
    loss  += (1.0f + 0.02f) * (accs[q] / (float)NZ);  // beta*mse + mse
    usage += accs[7 + q];
    perp  += accs[14 + q];
  }
  out_scalars[0] = usage / (float)QQ;
  out_scalars[1] = loss / (float)QQ;
  out_scalars[2] = perp / (float)QQ;
}

// ---------------------------------------------------------------------------
extern "C" void kernel_launch(void* const* d_in, const int* in_sizes, int n_in,
                              void* d_out, int out_size, void* d_ws,
                              size_t ws_size, hipStream_t stream) {
  (void)in_sizes; (void)n_in; (void)out_size; (void)ws_size;
  const float* z         = (const float*)d_in[0];
  const float* codebooks = (const float*)d_in[1];
  const float* phi_w     = (const float*)d_in[2];
  const float* phi_b     = (const float*)d_in[3];
  float* out = (float*)d_out;

  char* ws = (char*)d_ws;
  size_t off = 0;
  auto alloc = [&](size_t bytes) -> void* {
    off = (off + 255) & ~(size_t)255;
    void* p = ws + off;
    off += bytes;
    return p;
  };
  float*    zq      = (float*)alloc(NZ * 4);
  _Float16* cbh     = (_Float16*)alloc((size_t)QQ * VQ * CQ * 2);
  float*    cbnorm  = (float*)alloc((size_t)QQ * VQ * 4);
  _Float16* wre     = (_Float16*)alloc((size_t)QQ * 2 * 3 * CQ * CQ * 2);
  _Float16* tokh    = (_Float16*)alloc((size_t)BQ * TQ * CQ * 2);
  _Float16* ut      = (_Float16*)alloc((size_t)BQ * TP * CQ * 2);
  _Float16* h1      = (_Float16*)alloc((size_t)BQ * TP * CQ * 2);
  int*      idx_int = (int*)alloc((size_t)BQ * TQ * 4);
  int*      counts  = (int*)alloc((size_t)VQ * 4);
  float*    accs    = (float*)alloc(32 * 4);

  float* zhat     = out;                    // z_hat_out accumulated in place
  float* idx_base = out + NZ + 3;           // after usage,loss,perp scalars

  init_kernel<<<(int)(NZ / 256), 256, 0, stream>>>(z, zq, zhat);
  zero_scalars_kernel<<<1, 64, 0, stream>>>(accs);
  zero_pads_kernel<<<(BQ * CQ) / 256, 256, 0, stream>>>(ut, h1);
  cb_prep_kernel<<<QQ * VQ, 256, 0, stream>>>(codebooks, cbh, cbnorm);
  const size_t WTOT = (size_t)QQ * 2 * 3 * CQ * CQ;
  convert_w_kernel<<<(int)((WTOT + 255) / 256), 256, 0, stream>>>(phi_w, wre);

  static const int scales[QQ] = {1, 2, 4, 8, 16, 32, 64};
  int idxoff = 0;
  for (int q = 0; q < QQ; ++q) {
    const int s = scales[q];
    const int r = TQ / s;
    const int M = BQ * s;

    pool_tok_kernel<<<(M * CQ) / 256, 256, 0, stream>>>(zq, tokh, s, r);
    zero_counts_kernel<<<4, 256, 0, stream>>>(counts);
    score_argmax_kernel<<<M / 16, 256, 0, stream>>>(
        tokh, cbh + (size_t)q * VQ * CQ, cbnorm + q * VQ, idx_int,
        idx_base + idxoff, counts);
    gather_up_kernel<<<(int)(NZ / 256), 256, 0, stream>>>(
        codebooks + (size_t)q * VQ * CQ, idx_int, ut, s);
    conv_gemm_kernel<false><<<dim3(32, 32), 256, 0, stream>>>(
        wre + (size_t)(q * 2 + 0) * 3 * CQ * CQ, ut,
        phi_b + (size_t)(q * 2 + 0) * CQ, h1, nullptr, nullptr);
    conv_gemm_kernel<true><<<dim3(32, 32), 256, 0, stream>>>(
        wre + (size_t)(q * 2 + 1) * 3 * CQ * CQ, h1,
        phi_b + (size_t)(q * 2 + 1) * CQ, nullptr, zhat, zq);
    loss_reduce_kernel<<<2048, 256, 0, stream>>>(zhat, z, accs + q);
    perp_usage_kernel<<<1, 256, 0, stream>>>(counts, M, accs + 7 + q,
                                             accs + 14 + q);
    idxoff += M;
  }
  finalize_kernel<<<1, 1, 0, stream>>>(accs, out + NZ);
}
